// AutoRegressive_49555332661867
// MI455X (gfx1250) — matrix operand order repaired and checked by reading, not compiled, not run on hardware
//
#include <hip/hip_runtime.h>
#include <hip/hip_bf16.h>

// ---------------------------------------------------------------------------
// Problem constants (match reference)
// ---------------------------------------------------------------------------
constexpr int B_    = 512;
constexpr int T_    = 128;
constexpr int F_    = 64;
constexpr int U_    = 1024;   // UNITS
constexpr int OS_   = 32;     // OUT_STEPS
constexpr int NGATE = 4;
constexpr int KTOT  = F_ + U_;        // 1088 : combined K (x | h)
constexpr int ZCOLS = NGATE * U_;     // 4096

typedef __bf16 bf16;
typedef bf16  v16bf __attribute__((ext_vector_type(16)));
typedef bf16  v8bf  __attribute__((ext_vector_type(8)));
typedef float v8f   __attribute__((ext_vector_type(8)));

__device__ __forceinline__ float sigm_(float x) {
    return 1.0f / (1.0f + __expf(-x));
}
__device__ __forceinline__ float tanh_(float x) {
    return 2.0f / (1.0f + __expf(-2.0f * x)) - 1.0f;
}

// ---------------------------------------------------------------------------
// One-time preprocessing kernels
// ---------------------------------------------------------------------------
__global__ void k_cast_bf16(const float* __restrict__ in, bf16* __restrict__ out, int n) {
    int i = blockIdx.x * 256 + threadIdx.x;
    if (i < n) out[i] = (bf16)in[i];
}

// Wt[col][k] (col = gate*U + n, k in [0,1088)) built from Wk [64,4096], Wr [1024,4096]
__global__ void k_build_wt(const float* __restrict__ Wk, const float* __restrict__ Wr,
                           bf16* __restrict__ Wt) {
    int idx = blockIdx.x * 256 + threadIdx.x;
    if (idx >= ZCOLS * KTOT) return;
    int col = idx / KTOT;
    int k   = idx % KTOT;
    float v = (k < F_) ? Wk[(size_t)k * ZCOLS + col]
                       : Wr[(size_t)(k - F_) * ZCOLS + col];
    Wt[idx] = (bf16)v;
}

// WdT[n][k] (n in [0,64), k in [0,1024)) built from Wd [1024,64]
__global__ void k_build_wdt(const float* __restrict__ Wd, bf16* __restrict__ WdT) {
    int idx = blockIdx.x * 256 + threadIdx.x;
    if (idx >= F_ * U_) return;
    int n = idx / U_;
    int k = idx % U_;
    WdT[idx] = (bf16)Wd[(size_t)k * F_ + n];
}

__global__ void k_init_state(bf16* __restrict__ h0, float* __restrict__ c) {
    int i = blockIdx.x * 256 + threadIdx.x;
    if (i < B_ * U_) {
        h0[i] = (bf16)0.0f;
        c[i]  = 0.0f;
    }
}

// ---------------------------------------------------------------------------
// Fused LSTM step: z = [x|h] @ Wt(+b) for 4 gates, then cell update.
// Block = 256 threads = 8 waves laid out 4(M) x 2(N):
//   wave tile = 32 rows x 16 units x 4 gates  (8 WMMA accumulators)
//   block tile = 128 rows x 32 units
// grid = (U/32, B/128) = (32, 4)
// Per K-chunk per wave: 4 global_load_b128 (A) + 8 ds_load_b128 (B) + 8 WMMA
// K loop split into x-phase (2 chunks) and h-phase (32 chunks) so the hot
// h-phase uses pure pointer-offset addressing (no per-iter 64-bit muls).
// ---------------------------------------------------------------------------
__global__ __launch_bounds__(256)
void k_lstm_step(const bf16* __restrict__ xA, int xStride,     // x rows (bf16)
                 const bf16* __restrict__ hIn,                 // [B,U] bf16
                 const bf16* __restrict__ Wt,                  // [4096][1088] bf16
                 const float* __restrict__ bias,               // [4096] f32
                 float* __restrict__ cState,                   // [B,U] f32 (in/out)
                 bf16* __restrict__ hOut)                      // [B,U] bf16
{
    __shared__ alignas(128) bf16 ldsB[128 * 32];  // (4 gates x 32 units) x 32 k = 8 KB

    const int t      = threadIdx.x;
    const int lane   = t & 31;
    const int wave   = t >> 5;
    const int wN     = wave & 1;                       // N half within block
    const int wM     = wave >> 1;                      // M quarter within block
    const int nbBlk  = blockIdx.x * 32;                // block unit tile base
    const int nb     = nbBlk + wN * 16;                // wave unit tile base
    const int rowBase = blockIdx.y * 128 + wM * 32;    // wave batch tile base
    const int n      = lane & 15;                      // column within tile / A row
    const int kq     = lane >> 4;                      // K half select: k 16*kq..16*kq+15 (contiguous order used here)

    // B-tile cooperative staging: 128 cols x 32 k, 16 bf16 per thread
    const int cLoc  = t >> 1;                  // 0..127 : local column (gate*32 + unit)
    const int seg   = t & 1;                   // 0..1  : 16-element K segment
    const int gGate = cLoc >> 5;
    const int cN    = cLoc & 31;
    const bf16* wCur = Wt + (size_t)(gGate * U_ + nbBlk + cN) * KTOT + seg * 16;
    bf16* ldsStore   = &ldsB[cLoc * 32 + seg * 16];

    // B-fragment read base: column n of wave's 16-unit slice; +g*1024 per gate
    const bf16* ldsRead = &ldsB[(wN * 16 + n) * 32 + kq * 16];

    v8f acc[2][NGATE] = {};   // [m-subtile][gate], fp32 accumulators

    // A-row base pointers (lane supplies rows rowBase+n and rowBase+16+n)
    const bf16* aX0 = xA  + (size_t)(rowBase + n) * xStride + kq * 16;
    const bf16* aX1 = xA  + (size_t)(rowBase + 16 + n) * xStride + kq * 16;
    const bf16* aH0 = hIn + (size_t)(rowBase + n) * U_ + kq * 16;
    const bf16* aH1 = hIn + (size_t)(rowBase + 16 + n) * U_ + kq * 16;

    // one K-chunk: stage weights -> barrier -> A frags + 8 WMMAs
    auto chunk = [&](const bf16* a0, const bf16* a1, bool pf) {
        __syncthreads();
        *(v8bf*)(ldsStore)     = *(const v8bf*)(wCur);
        *(v8bf*)(ldsStore + 8) = *(const v8bf*)(wCur + 8);
        if (pf) __builtin_prefetch((const void*)(wCur + 32), 0, 3);
        __syncthreads();

        // A fragment layout: lanes 0-15 hold K {0..15}, lanes 16-31 {16..31}
        v8bf a0lo = *(const v8bf*)(a0);
        v8bf a0hi = *(const v8bf*)(a0 + 8);
        v8bf a1lo = *(const v8bf*)(a1);
        v8bf a1hi = *(const v8bf*)(a1 + 8);
        v16bf aF0 = __builtin_shufflevector(a0lo, a0hi,
                     0, 1, 2, 3, 4, 5, 6, 7, 8, 9, 10, 11, 12, 13, 14, 15);
        v16bf aF1 = __builtin_shufflevector(a1lo, a1hi,
                     0, 1, 2, 3, 4, 5, 6, 7, 8, 9, 10, 11, 12, 13, 14, 15);

        #pragma unroll
        for (int g = 0; g < NGATE; ++g) {
            v16bf bF = *(const v16bf*)(ldsRead + g * 32 * 32);
            acc[0][g] = __builtin_amdgcn_wmma_f32_16x16x32_bf16(
                            false, aF0, false, bF, (short)0, acc[0][g], false, false);
            acc[1][g] = __builtin_amdgcn_wmma_f32_16x16x32_bf16(
                            false, aF1, false, bF, (short)0, acc[1][g], false, false);
        }
        wCur += 32;
    };

    // x-phase: K = 0..63 (2 chunks)
    #pragma unroll
    for (int kc = 0; kc < F_; kc += 32) {
        chunk(aX0 + kc, aX1 + kc, true);
    }
    // h-phase: K = 64..1087 (32 chunks), pure pointer-offset addressing
    for (int kc = 0; kc < U_; kc += 32) {
        chunk(aH0 + kc, aH1 + kc, kc + 32 < U_);
    }

    // Epilogue: gates + cell update.  C/D layout: VGPR r, lane L -> (M = r + 8*(L>>4), N = L&15)
    const int col = nb + n;
    const float bi = bias[col];
    const float bf = bias[U_ + col];
    const float bg = bias[2 * U_ + col];
    const float bo = bias[3 * U_ + col];

    #pragma unroll
    for (int mt = 0; mt < 2; ++mt) {
        #pragma unroll
        for (int r = 0; r < 8; ++r) {
            const int row = rowBase + mt * 16 + r + 8 * kq;
            const size_t idx = (size_t)row * U_ + col;
            const float zi = acc[mt][0][r] + bi;
            const float zf = acc[mt][1][r] + bf;
            const float zg = acc[mt][2][r] + bg;
            const float zo = acc[mt][3][r] + bo;
            const float cNew = sigm_(zf) * cState[idx] + sigm_(zi) * tanh_(zg);
            cState[idx] = cNew;
            hOut[idx]   = (bf16)(sigm_(zo) * tanh_(cNew));
        }
    }
}

// ---------------------------------------------------------------------------
// Dense head: pred = h @ Wd + bd.  Writes f32 output slice + bf16 feedback.
// grid = (F/16, B/128), block = 256
// ---------------------------------------------------------------------------
__global__ __launch_bounds__(256)
void k_dense(const bf16* __restrict__ hIn,     // [B,U] bf16
             const bf16* __restrict__ WdT,     // [64][1024] bf16
             const float* __restrict__ bd,     // [64] f32
             float* __restrict__ out,          // [B, OS, F] f32
             int step,
             bf16* __restrict__ predB)         // [B, F] bf16
{
    const int t      = threadIdx.x;
    const int lane   = t & 31;
    const int wave   = t >> 5;
    const int nb     = blockIdx.x * 16;
    const int rowBase = blockIdx.y * 128 + wave * 16;
    const int n      = lane & 15;
    const int kq     = lane >> 4;

    v8f acc = {};
    const bf16* aRow = hIn + (size_t)(rowBase + n) * U_ + kq * 16;
    const bf16* bRow = WdT + (size_t)(nb + n) * U_ + kq * 16;

    for (int kc = 0; kc < U_; kc += 32) {
        v8bf alo = *(const v8bf*)(aRow + kc);
        v8bf ahi = *(const v8bf*)(aRow + kc + 8);
        v16bf aF = __builtin_shufflevector(alo, ahi,
                    0, 1, 2, 3, 4, 5, 6, 7, 8, 9, 10, 11, 12, 13, 14, 15);
        v16bf bF = *(const v16bf*)(bRow + kc);
        acc = __builtin_amdgcn_wmma_f32_16x16x32_bf16(
                  false, aF, false, bF, (short)0, acc, false, false);
    }

    const int col = nb + n;
    const float bb = bd[col];
    #pragma unroll
    for (int r = 0; r < 8; ++r) {
        const int row = rowBase + r + 8 * kq;
        const float p = acc[r] + bb;
        out[(size_t)row * (OS_ * F_) + (size_t)step * F_ + col] = p;
        predB[(size_t)row * F_ + col] = (bf16)p;
    }
}

// ---------------------------------------------------------------------------
// Host-side orchestration
// ---------------------------------------------------------------------------
extern "C" void kernel_launch(void* const* d_in, const int* in_sizes, int n_in,
                              void* d_out, int out_size, void* d_ws, size_t ws_size,
                              hipStream_t stream) {
    const float* inputs = (const float*)d_in[0];  // [B,T,F]
    const float* Wk     = (const float*)d_in[1];  // [64,4096]
    const float* Wr     = (const float*)d_in[2];  // [1024,4096]
    const float* b      = (const float*)d_in[3];  // [4096]
    const float* Wd     = (const float*)d_in[4];  // [1024,64]
    const float* bd     = (const float*)d_in[5];  // [64]
    float* out = (float*)d_out;                   // [B, OS, F]

    char* ws = (char*)d_ws;
    size_t off = 0;
    auto alloc = [&](size_t bytes) -> void* {
        void* p = ws + off;
        off = (off + bytes + 255) & ~(size_t)255;
        return p;
    };
    bf16* Wt    = (bf16*)alloc((size_t)ZCOLS * KTOT * sizeof(bf16));   // ~8.9 MB
    bf16* WdT   = (bf16*)alloc((size_t)F_ * U_ * sizeof(bf16));
    bf16* Xb    = (bf16*)alloc((size_t)B_ * T_ * F_ * sizeof(bf16));   // 8 MB
    bf16* hA    = (bf16*)alloc((size_t)B_ * U_ * sizeof(bf16));
    bf16* hB    = (bf16*)alloc((size_t)B_ * U_ * sizeof(bf16));
    float* cSt  = (float*)alloc((size_t)B_ * U_ * sizeof(float));
    bf16* predB = (bf16*)alloc((size_t)B_ * F_ * sizeof(bf16));

    // --- one-time preprocessing (deterministic, re-done each call) ---
    {
        int n = B_ * T_ * F_;
        k_cast_bf16<<<(n + 255) / 256, 256, 0, stream>>>(inputs, Xb, n);
        int nw = ZCOLS * KTOT;
        k_build_wt<<<(nw + 255) / 256, 256, 0, stream>>>(Wk, Wr, Wt);
        int nd = F_ * U_;
        k_build_wdt<<<(nd + 255) / 256, 256, 0, stream>>>(Wd, WdT);
        int ns = B_ * U_;
        k_init_state<<<(ns + 255) / 256, 256, 0, stream>>>(hA, cSt);
    }

    const dim3 gLstm(U_ / 32, B_ / 128);   // 32 x 4 blocks, 8 waves each
    const dim3 gDense(F_ / 16, B_ / 128);  // 4 x 4 blocks

    bf16* hbuf[2] = {hA, hB};
    int cur = 0;

    // warmup over the input window
    for (int t = 0; t < T_; ++t) {
        k_lstm_step<<<gLstm, 256, 0, stream>>>(Xb + (size_t)t * F_, T_ * F_,
                                               hbuf[cur], Wt, b, cSt, hbuf[cur ^ 1]);
        cur ^= 1;
    }
    // first prediction
    k_dense<<<gDense, 256, 0, stream>>>(hbuf[cur], WdT, bd, out, 0, predB);

    // autoregressive decode
    for (int s = 1; s < OS_; ++s) {
        k_lstm_step<<<gLstm, 256, 0, stream>>>(predB, F_,
                                               hbuf[cur], Wt, b, cSt, hbuf[cur ^ 1]);
        cur ^= 1;
        k_dense<<<gDense, 256, 0, stream>>>(hbuf[cur], WdT, bd, out, s, predB);
    }
}
